// IGConv_36429912605251
// MI455X (gfx1250) — compile-verified, matrix operand order checked by reading, not run on hardware
//
#include <hip/hip_runtime.h>

typedef __attribute__((ext_vector_type(16))) __bf16 v16bf;
typedef __attribute__((ext_vector_type(8)))  float  v8f;

#define D_FEAT 128   // feature dim
#define H_FEAT 128   // hidden dim

// ---------------- order-preserving f32 <-> u32 mapping (for atomic max) -------------
__device__ __forceinline__ unsigned f32_ord(float f) {
  unsigned u = __float_as_uint(f);
  return (u & 0x80000000u) ? ~u : (u | 0x80000000u);
}
__device__ __forceinline__ float ord_f32(unsigned u) {
  return (u & 0x80000000u) ? __uint_as_float(u & 0x7FFFFFFFu) : __uint_as_float(~u);
}
// enc(-inf) = ~0xFF800000 = 0x007FFFFF
#define ORD_NEG_INF 0x007FFFFFu

// ---------------- kernel 1: init aggregation buffers --------------------------------
__global__ void init_agg_kernel(unsigned* __restrict__ mx_enc, float* __restrict__ smbuf, int n) {
  int i = blockIdx.x * blockDim.x + threadIdx.x;
  if (i < n) { mx_enc[i] = ORD_NEG_INF; smbuf[i] = 0.0f; }
}

// ---------------- kernel 2: fold + convert weights to bf16 --------------------------
// W2a = W2[:, :128] + W2[:, 128:256] (mx appears twice in agg), W2b = W2[:, 256:384].
// W3p interleaves the z2/z1 halves of W3 column-wise: col 2k <- W3[:,k] (z2 part),
// col 2k+1 <- W3[:,128+k] (z1 part), matching the interleaved h staging layout.
__global__ void prep_weights_kernel(const float* __restrict__ W1,
                                    const float* __restrict__ W2,
                                    const float* __restrict__ W3,
                                    __bf16* __restrict__ W1bf,
                                    __bf16* __restrict__ W2abf,
                                    __bf16* __restrict__ W2bbf,
                                    __bf16* __restrict__ W3p) {
  int h = blockIdx.x;       // 0..127
  int k = threadIdx.x;      // 0..127
  W1bf[h * 128 + k]      = (__bf16)W1[h * 128 + k];
  W2abf[h * 128 + k]     = (__bf16)(W2[h * 384 + k] + W2[h * 384 + 128 + k]);
  W2bbf[h * 128 + k]     = (__bf16)W2[h * 384 + 256 + k];
  W3p[h * 256 + 2 * k]   = (__bf16)W3[h * 256 + k];
  W3p[h * 256 + 2 * k + 1] = (__bf16)W3[h * 256 + 128 + k];
}

// ---------------- kernel 3: edge gather + segment max/sum scatter -------------------
// one wave (32 lanes x float4 = 128 floats) per edge; indices scalarized so the
// gather is a single saddr b128 load; 4 f32 atomic-adds + 4 u32 atomic-maxes per lane.
__global__ void edge_scatter_kernel(const float* __restrict__ feat_src,
                                    const int* __restrict__ src_idx,
                                    const int* __restrict__ dst_idx,
                                    unsigned* __restrict__ mx_enc,
                                    float* __restrict__ smbuf,
                                    int n_edges) {
  const int t = blockIdx.x * blockDim.x + threadIdx.x;
  const int lane = t & 31;
  const int e = __builtin_amdgcn_readfirstlane(t >> 5);   // wave-uniform edge id
  if (e >= n_edges) return;
  const int s = __builtin_amdgcn_readfirstlane(src_idx[e]);
  const int d = __builtin_amdgcn_readfirstlane(dst_idx[e]);
  const float4 v = ((const float4*)(feat_src + (size_t)s * D_FEAT))[lane];
  float* sr = smbuf + (size_t)d * D_FEAT + lane * 4;
  atomicAdd(sr + 0, v.x); atomicAdd(sr + 1, v.y);
  atomicAdd(sr + 2, v.z); atomicAdd(sr + 3, v.w);
  unsigned* mr = mx_enc + (size_t)d * D_FEAT + lane * 4;
  atomicMax(mr + 0, f32_ord(v.x)); atomicMax(mr + 1, f32_ord(v.y));
  atomicMax(mr + 2, f32_ord(v.z)); atomicMax(mr + 3, f32_ord(v.w));
}

// ---------------- WMMA helpers ------------------------------------------------------
// B-fragment: B[k,n] = W[n,k]  =>  lane n reads row n of W. Per-lane K packing matches
// the ISA 16-bit layout: lane-half g: elements 0..7 -> K = 8g+e, 8..15 -> K = 16+8g+e.
__device__ __forceinline__ v16bf load_bfrag(const __bf16* __restrict__ W, int ldk,
                                            int n, int kk, int g) {
  const __bf16* p = W + (size_t)n * ldk + kk * 32 + g * 8;
  v16bf b;
#pragma unroll
  for (int e = 0; e < 8; ++e) { b[e] = p[e]; b[8 + e] = p[16 + e]; }
  return b;
}

struct BfPair { uint4 a, b; };   // 32B = one v16bf fragment

// ---------------- kernel 4: fused MLP (z2,z1 -> relu concat -> fc3 -> relu) ---------
__launch_bounds__(128)
__global__ void fused_mlp_kernel(const float* __restrict__ feat_dst,
                                 const unsigned* __restrict__ mx_enc,
                                 const float* __restrict__ smbuf,
                                 const __bf16* __restrict__ W1bf,
                                 const __bf16* __restrict__ W2abf,
                                 const __bf16* __restrict__ W2bbf,
                                 const __bf16* __restrict__ W3p,
                                 const float* __restrict__ b1,
                                 const float* __restrict__ b2,
                                 const float* __restrict__ b3,
                                 float* __restrict__ out, int n_dst) {
  // per-wave h staging, columns interleaved [z2_0, z1_0, z2_1, z1_1, ...] (32KB)
  __shared__ __align__(16) __bf16 hstage[4][16][256];
  const int wave = threadIdx.x >> 5;
  const int lane = threadIdx.x & 31;
  const int r0 = (blockIdx.x * 4 + wave) * 16;
  if (r0 >= n_dst) return;                 // uniform per wave -> EXEC stays all-1s
  const int rl = lane & 15;                // M row within tile / N col within tile
  const int g  = lane >> 4;                // lane half
  int row = r0 + rl;
  if (row > n_dst - 1) row = n_dst - 1;    // clamp loads, keep EXEC full for WMMA

  const unsigned* mrow = mx_enc   + (size_t)row * D_FEAT;
  const float*    srow = smbuf    + (size_t)row * D_FEAT;
  const float*    frow = feat_dst + (size_t)row * D_FEAT;

  // A-fragments for mx (decoded + zero-if-nonfinite), sum, feat_dst  (K=128 -> 4 frags)
  v16bf amx[4], asum[4], afd[4];
#pragma unroll
  for (int kk = 0; kk < 4; ++kk) {
#pragma unroll
    for (int hh = 0; hh < 2; ++hh) {
      const int kb = kk * 32 + hh * 16 + g * 8;
#pragma unroll
      for (int e = 0; e < 8; ++e) {
        float vm = ord_f32(mrow[kb + e]);
        if (!(vm >= -3.0e38f && vm <= 3.0e38f)) vm = 0.0f;  // -inf / NaN -> 0
        amx[kk][hh * 8 + e]  = (__bf16)vm;
        asum[kk][hh * 8 + e] = (__bf16)srow[kb + e];
        afd[kk][hh * 8 + e]  = (__bf16)frow[kb + e];
      }
    }
  }

  // stage 1: z2 = mx@W2a^T + sm@W2b^T + b2 ; z1 = fd@W1^T + b1 ; relu; pack into LDS
#pragma unroll 2
  for (int c = 0; c < 8; ++c) {
    v8f acc2 = {};
    v8f acc1 = {};
    const int n = c * 16 + rl;
#pragma unroll
    for (int kk = 0; kk < 4; ++kk)
      acc2 = __builtin_amdgcn_wmma_f32_16x16x32_bf16(false, amx[kk], false,
               load_bfrag(W2abf, 128, n, kk, g), (short)0, acc2, false, false);
#pragma unroll
    for (int kk = 0; kk < 4; ++kk)
      acc2 = __builtin_amdgcn_wmma_f32_16x16x32_bf16(false, asum[kk], false,
               load_bfrag(W2bbf, 128, n, kk, g), (short)0, acc2, false, false);
#pragma unroll
    for (int kk = 0; kk < 4; ++kk)
      acc1 = __builtin_amdgcn_wmma_f32_16x16x32_bf16(false, afd[kk], false,
               load_bfrag(W1bf, 128, n, kk, g), (short)0, acc1, false, false);
    const float bb2 = b2[n];
    const float bb1 = b1[n];
#pragma unroll
    for (int j = 0; j < 8; ++j) {          // D layout: lane half g -> M = 8g + j
      const int m = 8 * g + j;
      float z2 = acc2[j] + bb2; z2 = z2 > 0.0f ? z2 : 0.0f;
      float z1 = acc1[j] + bb1; z1 = z1 > 0.0f ? z1 : 0.0f;
      const unsigned short u2 = __builtin_bit_cast(unsigned short, (__bf16)z2);
      const unsigned short u1 = __builtin_bit_cast(unsigned short, (__bf16)z1);
      // interleaved columns: h[m][2n] = z2, h[m][2n+1] = z1  (single b32 store)
      ((unsigned*)&hstage[wave][m][0])[n] = (unsigned)u2 | ((unsigned)u1 << 16);
    }
  }

  // stage 2: repack h (16x256) from LDS into A-fragments via 16B ds loads
  v16bf ah[8];
  const __bf16* hrow = &hstage[wave][rl][0];
#pragma unroll
  for (int kk = 0; kk < 8; ++kk) {
    BfPair p;
    p.a = *(const uint4*)(hrow + kk * 32 + g * 8);        // K = 32kk + 8g .. +7
    p.b = *(const uint4*)(hrow + kk * 32 + 16 + g * 8);   // K = 32kk + 16 + 8g .. +7
    ah[kk] = __builtin_bit_cast(v16bf, p);
  }

  // stage 3: out = relu(h @ W3p^T + b3) with column-permuted W3p; predicated stores
#pragma unroll 2
  for (int c = 0; c < 8; ++c) {
    v8f acc = {};
    const int n = c * 16 + rl;
#pragma unroll
    for (int kk = 0; kk < 8; ++kk)
      acc = __builtin_amdgcn_wmma_f32_16x16x32_bf16(false, ah[kk], false,
              load_bfrag(W3p, 256, n, kk, g), (short)0, acc, false, false);
    const float bb3 = b3[n];
#pragma unroll
    for (int j = 0; j < 8; ++j) {
      const int m = r0 + 8 * g + j;
      float v = acc[j] + bb3; v = v > 0.0f ? v : 0.0f;
      if (m < n_dst) out[(size_t)m * H_FEAT + n] = v;
    }
  }
}

// ---------------- host-side launch ---------------------------------------------------
extern "C" void kernel_launch(void* const* d_in, const int* in_sizes, int n_in,
                              void* d_out, int out_size, void* d_ws, size_t ws_size,
                              hipStream_t stream) {
  const float* feat_src = (const float*)d_in[0];
  const float* feat_dst = (const float*)d_in[1];
  const float* W1 = (const float*)d_in[2];
  const float* b1 = (const float*)d_in[3];
  const float* W2 = (const float*)d_in[4];
  const float* b2 = (const float*)d_in[5];
  const float* W3 = (const float*)d_in[6];
  const float* b3 = (const float*)d_in[7];
  const int* src_idx = (const int*)d_in[8];
  const int* dst_idx = (const int*)d_in[9];

  const int n_dst   = in_sizes[1] / D_FEAT;
  const int n_edges = in_sizes[8];

  // workspace layout
  char* ws = (char*)d_ws;
  unsigned* mx_enc = (unsigned*)ws;                                   // n_dst*128 u32
  float*    smbuf  = (float*)(ws + (size_t)n_dst * D_FEAT * 4);       // n_dst*128 f32
  __bf16*   W1bf   = (__bf16*)(ws + (size_t)n_dst * D_FEAT * 8);
  __bf16*   W2abf  = W1bf  + 128 * 128;
  __bf16*   W2bbf  = W2abf + 128 * 128;
  __bf16*   W3p    = W2bbf + 128 * 128;                                // 128 x 256

  const int n_agg = n_dst * D_FEAT;
  init_agg_kernel<<<(n_agg + 255) / 256, 256, 0, stream>>>(mx_enc, smbuf, n_agg);
  prep_weights_kernel<<<128, 128, 0, stream>>>(W1, W2, W3, W1bf, W2abf, W2bbf, W3p);
  edge_scatter_kernel<<<(n_edges * 32 + 255) / 256, 256, 0, stream>>>(
      feat_src, src_idx, dst_idx, mx_enc, smbuf, n_edges);
  fused_mlp_kernel<<<(n_dst + 63) / 64, 128, 0, stream>>>(
      feat_dst, mx_enc, smbuf, W1bf, W2abf, W2bbf, W3p, b1, b2, b3,
      (float*)d_out, n_dst);
}
